// DGAttention_33440615367162
// MI455X (gfx1250) — matrix operand order repaired and checked by reading, not compiled, not run on hardware
//
#include <hip/hip_runtime.h>
#include <hip/hip_bf16.h>
#include <math.h>

// Problem constants (from reference): B=512 batch, D=256 dim, N=3 features.
#define BB 512
#define DD 256
#define LOG2E 1.44269504088896340736f

typedef __attribute__((ext_vector_type(16))) _Float16 v16h;
typedef __attribute__((ext_vector_type(8)))  _Float16 v8h;
typedef __attribute__((ext_vector_type(8)))  float    v8f;

// ---------------------------------------------------------------------------
// One-shot f32 -> f16 conversion (grid-stride). Inputs are L2-resident, so
// converting once avoids re-converting every tile inside the WMMA loop.
// ---------------------------------------------------------------------------
__global__ void cvt_f16_kernel(const float* __restrict__ src,
                               _Float16* __restrict__ dst, int n) {
    int idx = blockIdx.x * blockDim.x + threadIdx.x;
    int stride = gridDim.x * blockDim.x;
    for (; idx < n; idx += stride) dst[idx] = (_Float16)src[idx];
}

// ---------------------------------------------------------------------------
// Out[b][e] = relu( sum_d A[b][d] * W[e][d] + bias[e] ), f16 inputs, f32 out.
// A: [512 x 256] f16 row-major, W: [256 x 256] f16 row-major (e major = Wᵀ).
// One wave computes one 16x16 tile via v_wmma_f32_16x16x32_f16, K stepped 32.
// Fragment addressing (this code's contiguous k order, shared by A and B):
//   A (16x32 f16): lane(half,r) holds K=[k+16h,+16)
//   B (32x16 f16): B[d][e]=W[e][d] -> 16 contiguous halves of W row (n0+r)
//   C/D: VGPR v, lane -> [m0 + v + 8*half][n0 + r]
// Grid: (2, 32) x 256 threads; wave w handles n-tile blockIdx.x*8+w.
// ---------------------------------------------------------------------------
__global__ void gemm_relu_wmma(const _Float16* __restrict__ A,
                               const _Float16* __restrict__ W,
                               const float* __restrict__ bias,
                               float* __restrict__ Out) {
    const int lane = threadIdx.x & 31;
    const int wid  = threadIdx.x >> 5;
    const int half = lane >> 4;
    const int r    = lane & 15;

    const int m0 = blockIdx.y * 16;
    const int n0 = (blockIdx.x * 8 + wid) * 16;

    const _Float16* arow = A + (size_t)(m0 + r) * DD;
    const _Float16* wrow = W + (size_t)(n0 + r) * DD;

    v8f acc = {};
    for (int k = 0; k < DD; k += 32) {
        v16h af = *(const v16h*)(arow + k + 16 * half);
        v16h bf = *(const v16h*)(wrow + k + 16 * half);

        acc = __builtin_amdgcn_wmma_f32_16x16x32_f16(
            /*neg_a=*/false, af, /*neg_b=*/false, bf,
            /*c_mod=*/(short)0, acc, /*reuse_a=*/false, /*reuse_b=*/false);
    }

    const float bn = bias[n0 + r];
    #pragma unroll
    for (int v = 0; v < 8; ++v) {
        float val = acc[v] + bn;
        Out[(size_t)(m0 + v + 8 * half) * DD + (n0 + r)] = val > 0.f ? val : 0.f;
    }
}

// ---------------------------------------------------------------------------
// One block per (i, b). Each thread owns one p in [0,256).
// out[i][b][p] = (1/3)*gamma[i]*sum_j softmax_q(Q[i][j][b][p]*K[i][b][q]).V[i][b][:]
//               + F[i][b][p]
// Inner loop uses exp2 with K pre-scaled by log2(e) in LDS, float4 LDS reads.
// Per-(i,b) partial mean (sum_{j,p}Q * sum_q K) goes to bsums[i*BB+b]
// (deterministic; no atomics).
// ---------------------------------------------------------------------------
__global__ void attn_softmax_kernel(const float* __restrict__ Qbuf,
                                    const float* __restrict__ Kbuf,
                                    const float* __restrict__ Vbuf,
                                    const float* __restrict__ F0,
                                    const float* __restrict__ F1,
                                    const float* __restrict__ F2,
                                    const float* __restrict__ gamma,
                                    float* __restrict__ out,
                                    float* __restrict__ bsums) {
    const int i = blockIdx.x;
    const int b = blockIdx.y;
    const int p = threadIdx.x;

    __shared__ __align__(16) float Kld[DD];   // K * log2(e)
    __shared__ __align__(16) float Vld[DD];
    __shared__ float red[DD];

    const size_t rowOff = ((size_t)i * BB + b) * DD;
    const float kraw = Kbuf[rowOff + p];
    Kld[p] = kraw * LOG2E;
    Vld[p] = Vbuf[rowOff + p];

    // Block reductions over raw K row: max (softmax shift) and sum (means).
    red[p] = kraw;
    __syncthreads();
    for (int s = 128; s > 0; s >>= 1) {
        if (p < s) red[p] = fmaxf(red[p], red[p + s]);
        __syncthreads();
    }
    const float maxK2 = red[0] * LOG2E;
    __syncthreads();
    red[p] = kraw;
    __syncthreads();
    for (int s = 128; s > 0; s >>= 1) {
        if (p < s) red[p] += red[p + s];
        __syncthreads();
    }
    const float sumK = red[0];
    __syncthreads();

    const float4* K4 = (const float4*)Kld;
    const float4* V4 = (const float4*)Vld;

    float acc  = 0.f;
    float qtot = 0.f;
    #pragma unroll
    for (int j = 0; j < 3; ++j) {
        const float s = Qbuf[(((size_t)i * 3 + j) * BB + b) * DD + p];
        qtot += s;
        // Q,K are post-ReLU (>=0): max_q(s*K[q]) == s*maxK.
        const float m2 = (s >= 0.f) ? s * maxK2 : 0.f;
        float den = 0.f, num = 0.f;
        #pragma unroll 8
        for (int q = 0; q < DD / 4; ++q) {
            const float4 kq = K4[q];
            const float4 vq = V4[q];
            const float e0 = __builtin_amdgcn_exp2f(fmaf(s, kq.x, -m2));
            const float e1 = __builtin_amdgcn_exp2f(fmaf(s, kq.y, -m2));
            const float e2 = __builtin_amdgcn_exp2f(fmaf(s, kq.z, -m2));
            const float e3 = __builtin_amdgcn_exp2f(fmaf(s, kq.w, -m2));
            den += (e0 + e1) + (e2 + e3);
            num = fmaf(e0, vq.x, fmaf(e1, vq.y, fmaf(e2, vq.z, fmaf(e3, vq.w, num))));
        }
        acc += num / den;
    }

    // Deterministic partial for means[i]: (sum over j,p of Q) * (sum over q K)
    red[p] = qtot;
    __syncthreads();
    for (int s = 128; s > 0; s >>= 1) {
        if (p < s) red[p] += red[p + s];
        __syncthreads();
    }
    if (p == 0) bsums[(size_t)i * BB + b] = red[0] * sumK;

    const float* Fi = (i == 0) ? F0 : ((i == 1) ? F1 : F2);
    out[rowOff + p] = acc * (1.0f / 3.0f) * gamma[i] + Fi[(size_t)b * DD + p];
}

// ---------------------------------------------------------------------------
// Deterministic reduction of the 3 x 512 partial means, then 3-way softmax.
// ---------------------------------------------------------------------------
__global__ void alphas_kernel(const float* __restrict__ bsums,
                              float* __restrict__ alphas) {
    __shared__ float red[256];
    __shared__ float m[3];
    const int t = threadIdx.x;
    for (int i = 0; i < 3; ++i) {
        red[t] = bsums[(size_t)i * BB + t] + bsums[(size_t)i * BB + t + 256];
        __syncthreads();
        for (int s = 128; s > 0; s >>= 1) {
            if (t < s) red[t] += red[t + s];
            __syncthreads();
        }
        if (t == 0) m[i] = red[0];
        __syncthreads();
    }
    if (t == 0) {
        const float scale = 1.0f / (3.0f * (float)BB * (float)DD * (float)DD);
        float m0 = m[0] * scale, m1 = m[1] * scale, m2 = m[2] * scale;
        float mx = fmaxf(m0, fmaxf(m1, m2));
        float e0 = __expf(m0 - mx);
        float e1 = __expf(m1 - mx);
        float e2 = __expf(m2 - mx);
        float s  = e0 + e1 + e2;
        alphas[0] = e0 / s;
        alphas[1] = e1 / s;
        alphas[2] = e2 / s;
    }
}

extern "C" void kernel_launch(void* const* d_in, const int* in_sizes, int n_in,
                              void* d_out, int out_size, void* d_ws, size_t ws_size,
                              hipStream_t stream) {
    (void)in_sizes; (void)n_in; (void)out_size; (void)ws_size;

    const float* F[3] = {(const float*)d_in[0], (const float*)d_in[1],
                         (const float*)d_in[2]};
    const float* Wq    = (const float*)d_in[3];
    const float* bq    = (const float*)d_in[4];
    const float* Wk    = (const float*)d_in[5];
    const float* bk    = (const float*)d_in[6];
    const float* Wv    = (const float*)d_in[7];
    const float* bv    = (const float*)d_in[8];
    const float* gamma = (const float*)d_in[9];

    // Workspace layout:
    //   f32: Q [9][512][256], K [3][512][256], V [3][512][256], bsums [3][512]
    //   f16: Fh [3][512][256], Wqh/Wkh/Wvh [3][256][256] each
    // Total ~9.9 MB.
    float* ws    = (float*)d_ws;
    float* Qbuf  = ws;
    float* Kbuf  = Qbuf + (size_t)9 * BB * DD;
    float* Vbuf  = Kbuf + (size_t)3 * BB * DD;
    float* bsums = Vbuf + (size_t)3 * BB * DD;
    _Float16* Fh  = (_Float16*)(bsums + 3 * BB);
    _Float16* Wqh = Fh + (size_t)3 * BB * DD;
    _Float16* Wkh = Wqh + (size_t)3 * DD * DD;
    _Float16* Wvh = Wkh + (size_t)3 * DD * DD;

    // One-shot f32 -> f16 conversions.
    const int cvtThreads = 256;
    for (int j = 0; j < 3; ++j)
        cvt_f16_kernel<<<64, cvtThreads, 0, stream>>>(
            F[j], Fh + (size_t)j * BB * DD, BB * DD);
    cvt_f16_kernel<<<96, cvtThreads, 0, stream>>>(Wq, Wqh, 3 * DD * DD);
    cvt_f16_kernel<<<96, cvtThreads, 0, stream>>>(Wk, Wkh, 3 * DD * DD);
    cvt_f16_kernel<<<96, cvtThreads, 0, stream>>>(Wv, Wvh, 3 * DD * DD);

    const dim3 gGrid(2, 32);   // 16 n-tiles (8 waves x 2) x 32 m-tiles
    const dim3 gBlk(256);

    // K and V projections: params i applied to feat i.
    for (int i = 0; i < 3; ++i) {
        gemm_relu_wmma<<<gGrid, gBlk, 0, stream>>>(
            Fh + (size_t)i * BB * DD, Wkh + (size_t)i * DD * DD,
            bk + (size_t)i * DD, Kbuf + (size_t)i * BB * DD);
        gemm_relu_wmma<<<gGrid, gBlk, 0, stream>>>(
            Fh + (size_t)i * BB * DD, Wvh + (size_t)i * DD * DD,
            bv + (size_t)i * DD, Vbuf + (size_t)i * BB * DD);
    }
    // Q projections: params i applied to every feat j.
    for (int i = 0; i < 3; ++i) {
        for (int j = 0; j < 3; ++j) {
            gemm_relu_wmma<<<gGrid, gBlk, 0, stream>>>(
                Fh + (size_t)j * BB * DD, Wqh + (size_t)i * DD * DD,
                bq + (size_t)i * DD, Qbuf + ((size_t)i * 3 + j) * BB * DD);
        }
    }

    attn_softmax_kernel<<<dim3(3, BB), dim3(DD), 0, stream>>>(
        Qbuf, Kbuf, Vbuf, F[0], F[1], F[2], gamma, (float*)d_out, bsums);

    alphas_kernel<<<1, 256, 0, stream>>>(bsums,
                                         (float*)d_out + (size_t)3 * BB * DD);
}
